// clef_multimodal_51281909514850
// MI455X (gfx1250) — compile-verified, matrix operand order checked by reading, not run on hardware
//
#include <hip/hip_runtime.h>
#include <math.h>

// ---------------------------------------------------------------------------
// CDNA5 (gfx1250) wave32 WMMA implementation of the 2-layer transformer +
// pooling/MLP heads.  All big GEMMs run on v_wmma_f32_16x16x32_bf16 with
// pre-converted bf16 operands, double-buffered LDS tiles filled by
// global_load_async_to_lds_b128 (ASYNCcnt), fused flash attention on WMMA.
// ---------------------------------------------------------------------------

typedef __attribute__((ext_vector_type(16))) __bf16 v16bf;
typedef __attribute__((ext_vector_type(8)))  float  v8f;

#define B_   16
#define S_   1024
#define E_   1280
#define NH_  8
#define DH_  160
#define HID_ 128
#define BS_  (B_ * S_)

union bf4pack { __bf16 h[4]; uint2 u2; };

// ---- WMMA helpers ---------------------------------------------------------
// Fragment layout for 16-bit A (16x32) operands on wave32 (ISA 7.12.2):
//   lanes 0-15 : row = lane,    elems 0-7 = K[0..7],  elems 8-15 = K[16..23]
//   lanes16-31 : row = lane-16, elems 0-7 = K[8..15], elems 8-15 = K[24..31]
// B operands (32x16) use the same pattern with "row" = output column, so a
// K-contiguous (transposed for B) LDS tile serves both.
__device__ __forceinline__ v16bf ldfrag(const __bf16* base, int ld, int lane) {
  int r  = lane & 15;
  int kb = (lane >> 4) << 3;
  const __bf16* p = base + r * ld + kb;
  v16bf f;
#pragma unroll
  for (int i = 0; i < 8; ++i) f[i] = p[i];
#pragma unroll
  for (int i = 0; i < 8; ++i) f[8 + i] = p[16 + i];
  return f;
}

__device__ __forceinline__ v8f wmma_bf16(v16bf a, v16bf b, v8f c) {
  return __builtin_amdgcn_wmma_f32_16x16x32_bf16(
      /*neg_a=*/false, a, /*neg_b=*/false, b,
      /*c_mod=*/(short)0, c, /*reuse_a=*/false, /*reuse_b=*/false);
}

__device__ __forceinline__ void set8(v16bf& f, int off, float4 a, float4 b) {
  f[off + 0] = (__bf16)a.x; f[off + 1] = (__bf16)a.y;
  f[off + 2] = (__bf16)a.z; f[off + 3] = (__bf16)a.w;
  f[off + 4] = (__bf16)b.x; f[off + 5] = (__bf16)b.y;
  f[off + 6] = (__bf16)b.z; f[off + 7] = (__bf16)b.w;
}

// async copy: 16 bytes global -> LDS per lane, tracked by ASYNCcnt
__device__ __forceinline__ void async_ld16(unsigned lds_off, const __bf16* gptr) {
  asm volatile("global_load_async_to_lds_b128 %0, %1, off"
               :: "v"(lds_off), "v"(gptr) : "memory");
}

// ---------------------------------------------------------------------------
// GEMM:  C[M,N] = act( A[M,K] @ Wt[N,K]^T + bias )
// A, Wt are bf16 (Wt pre-transposed, K-contiguous).  Output f32 (Cf) or
// bf16 (Ch).  Block tile 128x128, BK=32, 8 waves (4x2), wave tile 32x64.
// Double-buffered LDS tiles filled with global_load_async_to_lds_b128.
// Requires N % 128 == 0, K % 32 == 0 (true at all call sites); M ragged
// (out-of-range rows clamp their source address; results never stored).
// ---------------------------------------------------------------------------
__global__ __launch_bounds__(256) void gemm_wmma_async(
    const __bf16* __restrict__ A, const __bf16* __restrict__ Wt,
    const float* __restrict__ bias, float* __restrict__ Cf,
    __bf16* __restrict__ Ch, int M, int N, int K, int relu)
{
  __shared__ __bf16 sA[2][128 * 40];   // [row][k], stride 40 (80B, 16B-aligned rows)
  __shared__ __bf16 sB[2][128 * 40];   // [col][k]

  int t    = threadIdx.x;
  int lane = t & 31;
  int wave = t >> 5;
  int wm   = wave >> 1;   // 0..3 -> 32-row slice
  int wn   = wave & 1;    // 0..1 -> 64-col slice
  int row0 = blockIdx.y * 128;
  int col0 = blockIdx.x * 128;

  // per-thread staging coordinates: 512 16B-chunks per tile, 2 per thread
  int rr[2], cc[2];
#pragma unroll
  for (int i = 0; i < 2; ++i) {
    int idx = i * 256 + t;
    rr[i] = idx >> 2;            // 0..127
    cc[i] = (idx & 3) << 3;      // 0,8,16,24
  }

  auto issue = [&](int k0, int buf) {
#pragma unroll
    for (int i = 0; i < 2; ++i) {
      int ar = row0 + rr[i]; if (ar >= M) ar = M - 1;   // clamp (never stored)
      async_ld16((unsigned)(size_t)&sA[buf][rr[i] * 40 + cc[i]],
                 A + (size_t)ar * K + k0 + cc[i]);
      async_ld16((unsigned)(size_t)&sB[buf][rr[i] * 40 + cc[i]],
                 Wt + (size_t)(col0 + rr[i]) * K + k0 + cc[i]);
    }
  };

  v8f acc[2][4];
#pragma unroll
  for (int i = 0; i < 2; ++i)
#pragma unroll
    for (int j = 0; j < 4; ++j)
#pragma unroll
      for (int r = 0; r < 8; ++r) acc[i][j][r] = 0.f;

  int nk = K >> 5;
  issue(0, 0);
  for (int ks = 0; ks < nk; ++ks) {
    int cur = ks & 1;
    if (ks + 1 < nk) {
      issue((ks + 1) << 5, cur ^ 1);                     // prefetch next tiles
      asm volatile("s_wait_asynccnt 0x4" ::: "memory");  // current batch done
    } else {
      asm volatile("s_wait_asynccnt 0x0" ::: "memory");
    }
    __syncthreads();

    v16bf af[2], bf_[4];
#pragma unroll
    for (int i = 0; i < 2; ++i)
      af[i] = ldfrag(&sA[cur][(wm * 32 + i * 16) * 40], 40, lane);
#pragma unroll
    for (int j = 0; j < 4; ++j)
      bf_[j] = ldfrag(&sB[cur][(wn * 64 + j * 16) * 40], 40, lane);
#pragma unroll
    for (int i = 0; i < 2; ++i)
#pragma unroll
      for (int j = 0; j < 4; ++j)
        acc[i][j] = wmma_bf16(af[i], bf_[j], acc[i][j]);
    __syncthreads();   // all reads of buf 'cur' done before it is refilled
  }

  // C-frag layout: VGPR r -> row (r + 8*(lane>=16)), col = lane&15 of tile
  int hi = lane >> 4, nn = lane & 15;
#pragma unroll
  for (int i = 0; i < 2; ++i)
#pragma unroll
    for (int j = 0; j < 4; ++j) {
      int col = col0 + wn * 64 + j * 16 + nn;
      float bv = bias ? bias[col] : 0.f;
#pragma unroll
      for (int r = 0; r < 8; ++r) {
        int row = row0 + wm * 32 + i * 16 + r + hi * 8;
        if (row < M) {
          float v = acc[i][j][r] + bv;
          if (relu) v = fmaxf(v, 0.f);
          if (Ch) Ch[(size_t)row * N + col] = (__bf16)v;
          else    Cf[(size_t)row * N + col] = v;
        }
      }
    }
}

// ---------------------------------------------------------------------------
// f32 -> bf16 stream conversion (n multiple of 4)
// ---------------------------------------------------------------------------
__global__ __launch_bounds__(256) void cvt_bf16_kernel(
    const float* __restrict__ in, __bf16* __restrict__ out, long n4)
{
  long i = (long)blockIdx.x * 256 + threadIdx.x;
  long stride = (long)gridDim.x * 256;
  for (; i < n4; i += stride) {
    float4 v = ((const float4*)in)[i];
    bf4pack p;
    p.h[0] = (__bf16)v.x; p.h[1] = (__bf16)v.y;
    p.h[2] = (__bf16)v.z; p.h[3] = (__bf16)v.w;
    ((uint2*)out)[i] = p.u2;
  }
}

// ---------------------------------------------------------------------------
// Transpose + convert:  Wt[n][k] = (bf16) W[k][n].  K,N multiples of 32.
// ---------------------------------------------------------------------------
__global__ __launch_bounds__(256) void transpose_cvt_kernel(
    const float* __restrict__ W, __bf16* __restrict__ Wt, int K, int N)
{
  __shared__ float tile[32][33];
  int tx = threadIdx.x & 31, ty = threadIdx.x >> 5;    // 32 x 8
  int n0 = blockIdx.x * 32, k0 = blockIdx.y * 32;
#pragma unroll
  for (int j = 0; j < 4; ++j)
    tile[ty * 4 + j][tx] = W[(size_t)(k0 + ty * 4 + j) * N + (n0 + tx)];
  __syncthreads();
#pragma unroll
  for (int j = 0; j < 4; ++j)
    Wt[(size_t)(n0 + ty * 4 + j) * K + (k0 + tx)] = (__bf16)tile[tx][ty * 4 + j];
}

// ---------------------------------------------------------------------------
// Fused attention (flash-style, online softmax).  One wave per
// (batch, head, 16-query tile).  DH=160 -> 5 K-chunks for QK^T and 10 N-tiles
// for PV.  Keys processed in chunks of 32 with key-validity masking.
// ---------------------------------------------------------------------------
__global__ __launch_bounds__(32) void attn_wmma(
    const float* __restrict__ q, const float* __restrict__ k,
    const float* __restrict__ v, const int* __restrict__ vlen,
    float* __restrict__ ctx)
{
  __shared__ __bf16 sV[DH_ * 32];  // [dh][key]  (transposed V chunk)
  __shared__ __bf16 sP[16 * 32];   // [qrow][key] softmax probs

  int lane = threadIdx.x;
  int qt = blockIdx.x, h = blockIdx.y, b = blockIdx.z;
  int vl = vlen[b] + 2;                 // attendable key count
  float scale = rsqrtf((float)DH_);
  int m  = lane & 15;
  int kb = (lane >> 4) << 3;
  int hi = lane >> 4;
  int nn = lane & 15;

  // Q tile as 5 bf16 A-fragments, pre-scaled by 1/sqrt(DH)
  v16bf qa[5];
  {
    const float* qrow = q + ((size_t)(b * S_ + qt * 16 + m)) * E_ + h * DH_;
#pragma unroll
    for (int d = 0; d < 5; ++d) {
      float4 a0 = *(const float4*)(qrow + d * 32 + kb);
      float4 a1 = *(const float4*)(qrow + d * 32 + kb + 4);
      float4 b0 = *(const float4*)(qrow + d * 32 + 16 + kb);
      float4 b1 = *(const float4*)(qrow + d * 32 + 16 + kb + 4);
      a0.x *= scale; a0.y *= scale; a0.z *= scale; a0.w *= scale;
      a1.x *= scale; a1.y *= scale; a1.z *= scale; a1.w *= scale;
      b0.x *= scale; b0.y *= scale; b0.z *= scale; b0.w *= scale;
      b1.x *= scale; b1.y *= scale; b1.z *= scale; b1.w *= scale;
      set8(qa[d], 0, a0, a1);
      set8(qa[d], 8, b0, b1);
    }
  }

  float rowm[8], rowl[8];
#pragma unroll
  for (int r = 0; r < 8; ++r) { rowm[r] = -1e30f; rowl[r] = 0.f; }
  v8f o[10];
#pragma unroll
  for (int j = 0; j < 10; ++j)
#pragma unroll
    for (int r = 0; r < 8; ++r) o[j][r] = 0.f;

  for (int c = 0; c < S_; c += 32) {
    // stage V chunk transposed: sV[dh][key]; lane <-> key
    {
      const float4* vr4 = (const float4*)(v + ((size_t)(b * S_ + c + lane)) * E_ + h * DH_);
#pragma unroll
      for (int d4 = 0; d4 < DH_ / 4; ++d4) {
        float4 x = vr4[d4];
        sV[(d4 * 4 + 0) * 32 + lane] = (__bf16)x.x;
        sV[(d4 * 4 + 1) * 32 + lane] = (__bf16)x.y;
        sV[(d4 * 4 + 2) * 32 + lane] = (__bf16)x.z;
        sV[(d4 * 4 + 3) * 32 + lane] = (__bf16)x.w;
      }
    }

    // scores for 2 x 16-key tiles; K memory rows already dh-contiguous
    v8f s0, s1;
#pragma unroll
    for (int r = 0; r < 8; ++r) { s0[r] = 0.f; s1[r] = 0.f; }
    const float* kr0 = k + ((size_t)(b * S_ + c + m)) * E_ + h * DH_;
    const float* kr1 = kr0 + (size_t)16 * E_;
#pragma unroll
    for (int d = 0; d < 5; ++d) {
      v16bf kf0, kf1;
      set8(kf0, 0, *(const float4*)(kr0 + d * 32 + kb),
                   *(const float4*)(kr0 + d * 32 + kb + 4));
      set8(kf0, 8, *(const float4*)(kr0 + d * 32 + 16 + kb),
                   *(const float4*)(kr0 + d * 32 + 16 + kb + 4));
      set8(kf1, 0, *(const float4*)(kr1 + d * 32 + kb),
                   *(const float4*)(kr1 + d * 32 + kb + 4));
      set8(kf1, 8, *(const float4*)(kr1 + d * 32 + 16 + kb),
                   *(const float4*)(kr1 + d * 32 + 16 + kb + 4));
      s0 = wmma_bf16(qa[d], kf0, s0);
      s1 = wmma_bf16(qa[d], kf1, s1);
    }

    bool ok0 = (c + nn) < vl;
    bool ok1 = (c + 16 + nn) < vl;
#pragma unroll
    for (int r = 0; r < 8; ++r) {
      if (!ok0) s0[r] = -1e30f;
      if (!ok1) s1[r] = -1e30f;
    }

    // online softmax: per-row max / sum across the 16 lanes of each half-wave
#pragma unroll
    for (int r = 0; r < 8; ++r) {
      float mx = fmaxf(s0[r], s1[r]);
      mx = fmaxf(mx, __shfl_xor(mx, 1, 16));
      mx = fmaxf(mx, __shfl_xor(mx, 2, 16));
      mx = fmaxf(mx, __shfl_xor(mx, 4, 16));
      mx = fmaxf(mx, __shfl_xor(mx, 8, 16));
      float mn = fmaxf(rowm[r], mx);
      float alpha = __expf(rowm[r] - mn);
      rowm[r] = mn;
      rowl[r] *= alpha;
#pragma unroll
      for (int j = 0; j < 10; ++j) o[j][r] *= alpha;
      float p0 = __expf(s0[r] - mn);
      float p1 = __expf(s1[r] - mn);
      sP[(r + hi * 8) * 32 + nn]      = (__bf16)p0;
      sP[(r + hi * 8) * 32 + 16 + nn] = (__bf16)p1;
      float ps = p0 + p1;
      ps += __shfl_xor(ps, 1, 16);
      ps += __shfl_xor(ps, 2, 16);
      ps += __shfl_xor(ps, 4, 16);
      ps += __shfl_xor(ps, 8, 16);
      rowl[r] += ps;
    }
    __syncthreads();

    // PV: A = P (16x32 keys), B = V chunk (32 keys x 160 dh) in 10 tiles
    v16bf pa = ldfrag(sP, 32, lane);
#pragma unroll
    for (int j = 0; j < 10; ++j) {
      v16bf vf = ldfrag(&sV[(j * 16) * 32], 32, lane);
      o[j] = wmma_bf16(pa, vf, o[j]);
    }
    __syncthreads();
  }

  // normalize and write ctx
#pragma unroll
  for (int r = 0; r < 8; ++r) {
    float inv = 1.f / rowl[r];
    int row = qt * 16 + r + hi * 8;
#pragma unroll
    for (int j = 0; j < 10; ++j)
      ctx[((size_t)(b * S_ + row)) * E_ + h * DH_ + j * 16 + nn] = o[j][r] * inv;
  }
}

// ---------------------------------------------------------------------------
// LayerNorm with optional residual:  out[row] = LN(x[row] + res[row]) * g + b
// ---------------------------------------------------------------------------
__global__ __launch_bounds__(256) void ln_kernel(
    const float* __restrict__ x, const float* __restrict__ res,
    const float* __restrict__ g, const float* __restrict__ bta,
    float* __restrict__ out, int width)
{
  int row = blockIdx.x;
  int t = threadIdx.x;
  const float* xr = x + (size_t)row * width;
  const float* rr = res ? res + (size_t)row * width : nullptr;
  __shared__ float s1[256], s2[256];

  float sum = 0.f, sq = 0.f;
  for (int c = t; c < width; c += 256) {
    float v = xr[c] + (rr ? rr[c] : 0.f);
    sum += v; sq += v * v;
  }
  s1[t] = sum; s2[t] = sq;
  __syncthreads();
  for (int off = 128; off > 0; off >>= 1) {
    if (t < off) { s1[t] += s1[t + off]; s2[t] += s2[t + off]; }
    __syncthreads();
  }
  float mean = s1[0] / (float)width;
  float var  = s2[0] / (float)width - mean * mean;
  float rstd = rsqrtf(var + 1e-5f);
  for (int c = t; c < width; c += 256) {
    float v = xr[c] + (rr ? rr[c] : 0.f);
    out[(size_t)row * width + c] = (v - mean) * rstd * g[c] + bta[c];
  }
}

// ---------------------------------------------------------------------------
// Ragged masked mean over positions [0, vl+2):  pooled[b,e]
// ---------------------------------------------------------------------------
__global__ __launch_bounds__(256) void pool_kernel(
    const float* __restrict__ x, const int* __restrict__ vlen,
    float* __restrict__ pooled)
{
  int b = blockIdx.x;
  int n = vlen[b] + 2;
  float inv = 1.f / (float)n;
  for (int e = threadIdx.x; e < E_; e += 256) {
    float s = 0.f;
    for (int ss = 0; ss < n; ++ss)
      s += x[((size_t)(b * S_ + ss)) * E_ + e];
    pooled[b * E_ + e] = s * inv;
  }
}

// ---------------------------------------------------------------------------
// Standardize modal_a/modal_b over batch dim (ddof=1) and concat -> cross
// ---------------------------------------------------------------------------
__global__ __launch_bounds__(256) void standardize_kernel(
    const float* __restrict__ a, const float* __restrict__ bmat,
    float* __restrict__ cross)
{
  int j = blockIdx.x * blockDim.x + threadIdx.x;   // 0..1023
  if (j >= 1024) return;
  const float* src = (j < 512) ? (a + j) : (bmat + (j - 512));
  float mean = 0.f;
#pragma unroll
  for (int i = 0; i < B_; ++i) mean += src[i * 512];
  mean *= (1.f / (float)B_);
  float var = 0.f;
#pragma unroll
  for (int i = 0; i < B_; ++i) {
    float d = src[i * 512] - mean;
    var += d * d;
  }
  var *= (1.f / (float)(B_ - 1));
  float s = sqrtf(var);
#pragma unroll
  for (int i = 0; i < B_; ++i)
    cross[i * 1024 + j] = (src[i * 512] - mean) / s;
}

// ---------------------------------------------------------------------------
static inline void run_cvt(const float* in, __bf16* out, long n, hipStream_t s) {
  long n4 = n >> 2;
  int blocks = (int)((n4 + 255) / 256);
  if (blocks > 4096) blocks = 4096;
  cvt_bf16_kernel<<<blocks, 256, 0, s>>>(in, out, n4);
}
static inline void run_tcvt(const float* W, __bf16* Wt, int K, int N, hipStream_t s) {
  transpose_cvt_kernel<<<dim3(N / 32, K / 32), 256, 0, s>>>(W, Wt, K, N);
}
static inline void run_gemm(const __bf16* A, const __bf16* Wt, const float* bias,
                            float* Cf, __bf16* Ch, int M, int N, int K, int relu,
                            hipStream_t s) {
  dim3 grid(N / 128, (M + 127) / 128);
  gemm_wmma_async<<<grid, 256, 0, s>>>(A, Wt, bias, Cf, Ch, M, N, K, relu);
}

extern "C" void kernel_launch(void* const* d_in, const int* in_sizes, int n_in,
                              void* d_out, int out_size, void* d_ws, size_t ws_size,
                              hipStream_t stream) {
  (void)in_sizes; (void)n_in; (void)out_size; (void)ws_size;

  const float* esm     = (const float*)d_in[0];
  const float* modal_a = (const float*)d_in[1];
  const float* modal_b = (const float*)d_in[2];
  const float* Wq = (const float*)d_in[3];  const float* bq = (const float*)d_in[4];
  const float* Wk = (const float*)d_in[5];  const float* bk = (const float*)d_in[6];
  const float* Wv = (const float*)d_in[7];  const float* bv = (const float*)d_in[8];
  const float* Wo = (const float*)d_in[9];  const float* bo = (const float*)d_in[10];
  const float* ln1g = (const float*)d_in[11]; const float* ln1b = (const float*)d_in[12];
  const float* W1 = (const float*)d_in[13]; const float* b1 = (const float*)d_in[14];
  const float* W2 = (const float*)d_in[15]; const float* b2 = (const float*)d_in[16];
  const float* ln2g = (const float*)d_in[17]; const float* ln2b = (const float*)d_in[18];
  const float* mlpW1 = (const float*)d_in[19]; const float* mlpb1 = (const float*)d_in[20];
  const float* mlpW2 = (const float*)d_in[21]; const float* mlpb2 = (const float*)d_in[22];
  const float* featW1 = (const float*)d_in[23];
  const float* featW2 = (const float*)d_in[24]; const float* featb2 = (const float*)d_in[25];
  const float* lnfg = (const float*)d_in[26]; const float* lnfb = (const float*)d_in[27];
  const int*   vlen = (const int*)d_in[28];

  float* out = (float*)d_out;
  float* pooled = out;                  // [16,1280]
  float* projX  = out + B_ * E_;        // [16,128]
  float* pf     = projX + B_ * HID_;    // [16,128]

  // ---- workspace layout (floats unless noted) ----
  float* ws = (float*)d_ws;
  const size_t nXE = (size_t)BS_ * E_;                 // 20,971,520
  float*  xbuf = ws;                                   // [BS,E] f32
  float*  tmp  = ws + nXE;                             // [BS,E] f32 scratch
  float*  big  = ws + 2 * nXE;                         // 4*nXE f32 region
  float*  qf   = big;                                  // [BS,E]
  float*  kf   = big + nXE;                            // [BS,E]
  float*  vf   = big + 2 * nXE;                        // [BS,E]
  float*  obuf = big + 3 * nXE;                        // [BS,E]
  __bf16* hbf  = (__bf16*)big;                         // [BS,4E] bf16 (reuses q/k/v/o)
  __bf16* xbf  = (__bf16*)(ws + 6 * nXE);              // [BS,E] bf16 activation
  __bf16* wt   = (__bf16*)(ws + 6 * nXE + nXE / 2);    // transposed weights, <=6.6M bf16
  float*  smallf = ws + 6 * nXE + nXE / 2 + 3500000;   // small f32 region (16B aligned)
  float*  pftmp  = smallf;                             // [16,128]
  float*  cross  = smallf + 2048;                      // [16,1024]
  __bf16* sbf    = (__bf16*)(smallf + 2048 + 16384);   // [16,1280] bf16
  __bf16* h1bf   = sbf + B_ * E_;                      // [16,2560] bf16
  __bf16* crossbf = h1bf + (size_t)B_ * 2 * E_;        // [16,1024] bf16

  for (int l = 0; l < 2; ++l) {
    const float* xin = (l == 0) ? esm : xbuf;
    const size_t wE2 = (size_t)l * E_ * E_;
    // bf16 activation for QKV
    run_cvt(xin, xbf, (long)nXE, stream);
    // QKV projections (f32 out for attention)
    run_tcvt(Wq + wE2, wt, E_, E_, stream);
    run_gemm(xbf, wt, bq + (size_t)l * E_, qf, nullptr, BS_, E_, E_, 0, stream);
    run_tcvt(Wk + wE2, wt, E_, E_, stream);
    run_gemm(xbf, wt, bk + (size_t)l * E_, kf, nullptr, BS_, E_, E_, 0, stream);
    run_tcvt(Wv + wE2, wt, E_, E_, stream);
    run_gemm(xbf, wt, bv + (size_t)l * E_, vf, nullptr, BS_, E_, E_, 0, stream);
    // fused attention -> ctx in tmp
    attn_wmma<<<dim3(S_ / 16, NH_, B_), 32, 0, stream>>>(qf, kf, vf, vlen, tmp);
    // output projection + residual LN
    run_cvt(tmp, xbf, (long)nXE, stream);
    run_tcvt(Wo + wE2, wt, E_, E_, stream);
    run_gemm(xbf, wt, bo + (size_t)l * E_, obuf, nullptr, BS_, E_, E_, 0, stream);
    ln_kernel<<<BS_, 256, 0, stream>>>(xin, obuf, ln1g + (size_t)l * E_,
                                       ln1b + (size_t)l * E_, xbuf, E_);
    // FFN: hidden produced directly in bf16
    run_cvt(xbuf, xbf, (long)nXE, stream);
    run_tcvt(W1 + (size_t)l * E_ * 4 * E_, wt, E_, 4 * E_, stream);
    run_gemm(xbf, wt, b1 + (size_t)l * 4 * E_, nullptr, hbf, BS_, 4 * E_, E_, 1, stream);
    run_tcvt(W2 + (size_t)l * 4 * E_ * E_, wt, 4 * E_, E_, stream);
    run_gemm(hbf, wt, b2 + (size_t)l * E_, tmp, nullptr, BS_, E_, 4 * E_, 0, stream);
    ln_kernel<<<BS_, 256, 0, stream>>>(xbuf, tmp, ln2g + (size_t)l * E_,
                                       ln2b + (size_t)l * E_, xbuf, E_);
  }

  // ragged mean pooling -> pooled (output 0)
  pool_kernel<<<B_, 256, 0, stream>>>(xbuf, vlen, pooled);

  // proj_X = relu(pooled @ mlp_W1 + b1) @ mlp_W2 + b2  (output 1)
  run_cvt(pooled, sbf, B_ * E_, stream);
  run_tcvt(mlpW1, wt, E_, 2 * E_, stream);
  run_gemm(sbf, wt, mlpb1, nullptr, h1bf, B_, 2 * E_, E_, 1, stream);
  run_tcvt(mlpW2, wt, 2 * E_, HID_, stream);
  run_gemm(h1bf, wt, mlpb2, projX, nullptr, B_, HID_, 2 * E_, 0, stream);

  // pf = LN( relu(standardize(a)||standardize(b) @ feat_W1) @ feat_W2 + b2 )
  standardize_kernel<<<4, 256, 0, stream>>>(modal_a, modal_b, cross);
  run_cvt(cross, crossbf, B_ * 1024, stream);
  run_tcvt(featW1, wt, 1024, 1024, stream);
  run_gemm(crossbf, wt, nullptr, nullptr, h1bf, B_, 1024, 1024, 1, stream);
  run_tcvt(featW2, wt, 1024, HID_, stream);
  run_gemm(h1bf, wt, featb2, pftmp, nullptr, B_, HID_, 1024, 0, stream);
  ln_kernel<<<B_, 256, 0, stream>>>(pftmp, nullptr, lnfg, lnfb, pf, HID_);
}
